// Attention_22677427323029
// MI455X (gfx1250) — compile-verified, matrix operand order checked by reading, not run on hardware
//
#include <hip/hip_runtime.h>
#include <hip/hip_bf16.h>

// ---------------------------------------------------------------------------
// MI455X (gfx1250) attention pipeline — wave32, v_wmma_f32_16x16x32_f16,
// async global->LDS staging (ASYNCcnt) with double-buffered LDS tiles.
//   0) cvt_f32_f16     : x, Wq..Wo -> f16 (once)
//   1) gemm16          : Q/K/V projections (f16 WMMA, f32 accum)
//   2) rope_pack       : RoPE + [B,nh,S,hd] transpose + f16 pack (+ k,v out)
//   3) transpose_hd_seq: V -> [bh][hd][S] so PV B-fragments are contiguous
//   4) flash_attn      : causal flash attention (all matmuls on v_wmma)
//   5) gemm16          : output projection -> y (fp32)
// ---------------------------------------------------------------------------

#define USE_ASYNC_LDS 1

typedef __attribute__((ext_vector_type(16))) _Float16 half16;
typedef __attribute__((ext_vector_type(8)))  _Float16 half8;
typedef __attribute__((ext_vector_type(4)))  _Float16 half4;
typedef __attribute__((ext_vector_type(8)))  float    float8;

#define WMMA_F32_F16(a, b, c) \
  __builtin_amdgcn_wmma_f32_16x16x32_f16(false, (a), false, (b), (short)0, (c), false, false)

// Problem constants
#define BATCH 2
#define SEQ   2048
#define HID   1024
#define NH    16
#define HD    64
#define MROWS (BATCH * SEQ)          // 4096
#define PLANE ((size_t)MROWS * HID)  // 4,194,304 elements

// ---------------------------------------------------------------------------
// 32-byte global->LDS copy. Async path: per ISA §15.18.3 the instruction
// copies 16 bytes global->LDS; ioffset applies to BOTH
// addresses, so the 2nd chunk reuses the same operands with offset:16.
// LDS byte offset = low 32 bits of the generic (flat) address (§10.2).
// Completion is tracked with ASYNCcnt -> s_wait_asynccnt.
// ---------------------------------------------------------------------------
__device__ __forceinline__ void lds_cp32(void* lds_dst, const void* gsrc) {
#if USE_ASYNC_LDS
  const unsigned off = (unsigned)(unsigned long long)lds_dst;
  asm volatile("global_load_async_to_lds_b128 %0, %1, off\n\t"
               "global_load_async_to_lds_b128 %0, %1, off offset:16"
               :: "v"(off), "v"(gsrc)
               : "memory");
#else
  ((half8*)lds_dst)[0] = ((const half8*)gsrc)[0];
  ((half8*)lds_dst)[1] = ((const half8*)gsrc)[1];
#endif
}

#if USE_ASYNC_LDS
// 4 async copies are issued per tile per wave; waiting to <=4 means "current
// tile complete, next tile still in flight".
#define ASYNC_WAIT_TILE() asm volatile("s_wait_asynccnt 4" ::: "memory")
#else
#define ASYNC_WAIT_TILE()
#endif

// ---------------------------------------------------------------------------
// Fragment loader: 16 halves per lane (ISA 7.12.2 16-bit A layout):
//   lane&15 = row (A) / col (B), koff = kchunk + (lane>>4)*8,
//   halves = K[koff..koff+7] ++ K[koff+16..koff+23].
// ld*2 must be a multiple of 16 bytes so both v8h loads stay 16B-aligned.
// ---------------------------------------------------------------------------
__device__ __forceinline__ half16 frag_ld(const _Float16* __restrict__ base,
                                          int ld, int lane, int kchunk) {
  const int r    = lane & 15;
  const int koff = kchunk + ((lane >> 4) << 3);
  const _Float16* p = base + r * ld + koff;
  const half8 lo = *(const half8*)p;
  const half8 hi = *(const half8*)(p + 16);
  half16 f;
#pragma unroll
  for (int i = 0; i < 8; ++i) { f[i] = lo[i]; f[i + 8] = hi[i]; }
  return f;
}

// ---------------------------------------------------------------------------
// fp32 -> f16 conversion, 4 elements per thread.
// ---------------------------------------------------------------------------
__global__ __launch_bounds__(256) void cvt_f32_f16(const float* __restrict__ s,
                                                   _Float16* __restrict__ d) {
  const size_t i = ((size_t)blockIdx.x * 256 + threadIdx.x) * 4;
  const float4 f = *(const float4*)(s + i);
  half4 h;
  h[0] = (_Float16)f.x; h[1] = (_Float16)f.y;
  h[2] = (_Float16)f.z; h[3] = (_Float16)f.w;
  *(half4*)(d + i) = h;
}

// ---------------------------------------------------------------------------
// GEMM: C[M,N] = A[M,K] * W[N,K]^T  (f16 in, f32 out)
// Block 128x128, BK=32, 8 waves as 4x2, wave tile 32x64 (2x4 WMMA tiles).
// Double-buffered LDS; async staging always issued one K-step ahead with the
// prefetch index clamped (straight-line inner loop, no branches).
// ---------------------------------------------------------------------------
#define GBM 128
#define GBN 128
#define GBK 32
#define GLD 48  // 32 + 16 halves pad; 96B row stride (16B multiple)

__global__ __launch_bounds__(256) void gemm16(const _Float16* __restrict__ A,
                                              const _Float16* __restrict__ W,
                                              float* __restrict__ C,
                                              int M, int N, int K) {
  __shared__ _Float16 As[2][GBM][GLD];   // 2 x 12 KB
  __shared__ _Float16 Ws[2][GBN][GLD];   // 2 x 12 KB

  const int tid  = threadIdx.x;
  const int lane = tid & 31;
  const int wave = tid >> 5;
  const int wm   = wave >> 1;   // 0..3
  const int wn   = wave & 1;    // 0..1
  const int m0   = blockIdx.y * GBM;
  const int n0   = blockIdx.x * GBN;

  const int sr = tid >> 1;        // staging row 0..127
  const int sc = (tid & 1) << 4;  // staging col {0,16} halves

  const _Float16* gA = A + (size_t)(m0 + sr) * K + sc;
  const _Float16* gW = W + (size_t)(n0 + sr) * K + sc;

  float8 acc[2][4] = {};

  lds_cp32(&As[0][sr][sc], gA);   // stage K-step 0
  lds_cp32(&Ws[0][sr][sc], gW);

  const int nk = K / GBK;
  for (int kt = 0; kt < nk; ++kt) {
    const int bi = kt & 1;
    // Always stage "next" (clamped: last iteration restages the final tile
    // into the idle buffer; nothing reads it and s_endpgm drains ASYNCcnt).
    const int knext = (kt + 1 < nk) ? (kt + 1) * GBK : kt * GBK;
    lds_cp32(&As[bi ^ 1][sr][sc], gA + knext);
    lds_cp32(&Ws[bi ^ 1][sr][sc], gW + knext);
    ASYNC_WAIT_TILE();            // current tile done, next stays in flight
    __syncthreads();

    const half16 a0 = frag_ld(&As[bi][wm * 32][0],      GLD, lane, 0);
    const half16 a1 = frag_ld(&As[bi][wm * 32 + 16][0], GLD, lane, 0);
#pragma unroll
    for (int j = 0; j < 4; ++j) {
      const half16 b = frag_ld(&Ws[bi][wn * 64 + j * 16][0], GLD, lane, 0);
      acc[0][j] = WMMA_F32_F16(a0, b, acc[0][j]);
      acc[1][j] = WMMA_F32_F16(a1, b, acc[1][j]);
    }
    __syncthreads();
  }

  // C/D layout: lane&15 = n, VGPR v -> m = v + (lane>>4)*8
#pragma unroll
  for (int i = 0; i < 2; ++i)
#pragma unroll
    for (int j = 0; j < 4; ++j) {
      const int mb = m0 + wm * 32 + i * 16 + ((lane >> 4) << 3);
      const int nn = n0 + wn * 64 + j * 16 + (lane & 15);
#pragma unroll
      for (int v = 0; v < 8; ++v)
        C[(size_t)(mb + v) * N + nn] = acc[i][j][v];
    }
}

// ---------------------------------------------------------------------------
// RoPE + head-major transpose + f16 pack.  One thread per (b,s,head,d).
// q16 is pre-scaled by 1/sqrt(hd); k/v also written fp32 (they are outputs).
// ---------------------------------------------------------------------------
__global__ __launch_bounds__(256) void rope_pack(
    const float* __restrict__ qf, const float* __restrict__ kf,
    const float* __restrict__ vf, const int* __restrict__ pos_ids,
    _Float16* __restrict__ q16, _Float16* __restrict__ k16,
    _Float16* __restrict__ v16, float* __restrict__ k_out,
    float* __restrict__ v_out) {
  const int idx  = blockIdx.x * 256 + threadIdx.x;  // [0, B*S*NH*HD)
  const int d    = idx & 63;
  const int head = (idx >> 6) & 15;
  const int s    = (idx >> 10) & 2047;
  const int b    = idx >> 21;

  const int row = b * SEQ + s;
  const int src = row * HID + head * HD + d;

  const float pos = (float)pos_ids[row];
  const int   dr  = d & 31;
  // inv_freq[dr] = 10000^(-dr/32) = exp(-dr * ln(10000)/32)
  const float freq = pos * __expf(-(float)dr * 0.28782313662f);
  float sn, cs;
  __sincosf(freq, &sn, &cs);

  const float q = qf[src], k = kf[src], v = vf[src];
  const float qrot = (d < 32) ? -qf[src + 32] : qf[src - 32];
  const float krot = (d < 32) ? -kf[src + 32] : kf[src - 32];
  const float qo = q * cs + qrot * sn;
  const float ko = k * cs + krot * sn;

  const int dst = ((b * NH + head) * SEQ + s) * HD + d;  // [B,nh,S,hd]
  q16[dst]  = (_Float16)(qo * 0.125f);  // fold 1/sqrt(64) into Q
  k16[dst]  = (_Float16)ko;
  v16[dst]  = (_Float16)v;
  k_out[dst] = ko;   // present_k output [B,nh,S,hd]
  v_out[dst] = v;    // present_v output
}

// ---------------------------------------------------------------------------
// V transpose: [bh][S][hd] -> [bh][hd][S] (64x64 LDS tiles, both sides
// coalesced) so flash-attention's PV B-fragments are contiguous row reads.
// ---------------------------------------------------------------------------
__global__ __launch_bounds__(256) void transpose_hd_seq(
    const _Float16* __restrict__ src, _Float16* __restrict__ dst) {
  __shared__ _Float16 T[64][72];
  const int bh = blockIdx.y;
  const int s0 = blockIdx.x * 64;
  const int r  = threadIdx.x >> 2;         // 0..63
  const int c  = (threadIdx.x & 3) << 4;   // 0,16,32,48

  const _Float16* sp = src + ((size_t)bh * SEQ + s0 + r) * HD + c;
  *(half8*)&T[r][c]     = *(const half8*)sp;
  *(half8*)&T[r][c + 8] = *(const half8*)(sp + 8);
  __syncthreads();

  half8 o0, o1;
#pragma unroll
  for (int i = 0; i < 8; ++i) { o0[i] = T[c + i][r]; o1[i] = T[c + 8 + i][r]; }
  _Float16* dp = dst + ((size_t)bh * HD + r) * SEQ + s0 + c;
  *(half8*)dp       = o0;
  *(half8*)(dp + 8) = o1;
}

// ---------------------------------------------------------------------------
// Causal flash attention. Block = (b,h, 128 Q rows); 8 waves x 16 rows each.
// Double-buffered K/V tiles staged with async copies one tile ahead
// (clamped prefetch index -> branch-free inner loop).
// Writes f16 directly (feeds the output-projection GEMM).
// ---------------------------------------------------------------------------
#define FLD 80  // 64 + 16 halves pad; 160B row stride (16B multiple)

__global__ __launch_bounds__(256) void flash_attn(
    const _Float16* __restrict__ q16, const _Float16* __restrict__ k16,
    const _Float16* __restrict__ v16t,  // [bh][hd][S]
    _Float16* __restrict__ attn16) {    // [B,S,H]
  __shared__ _Float16 Ks[2][64][FLD];   // [kpos][d]      2 x 10 KB
  __shared__ _Float16 Vt[2][64][FLD];   // [d][kpos]      2 x 10 KB
  __shared__ _Float16 Ps[8][16][FLD];   // per-wave P     20 KB

  const int tid   = threadIdx.x;
  const int lane  = tid & 31;
  const int wave  = tid >> 5;
  const int bh    = blockIdx.y;         // b*NH + h
  const int qbase = blockIdx.x * 128;
  const int b     = bh >> 4;
  const int h     = bh & 15;
  const size_t plane = (size_t)bh * SEQ * HD;

  const int qrow0 = qbase + wave * 16;
  const _Float16* qp = q16 + plane + (size_t)qrow0 * HD;
  const half16 qf0 = frag_ld(qp, HD, lane, 0);
  const half16 qf1 = frag_ld(qp, HD, lane, 32);

  float mrow[8], lrow[8];
  float8 o[4] = {};
#pragma unroll
  for (int v = 0; v < 8; ++v) { mrow[v] = -1e30f; lrow[v] = 0.0f; }

  const int nkt = qbase / 64 + 2;  // K tiles covering [0, qbase+128)
  const int srr = tid >> 2;        // staging row 0..63
  const int scc = (tid & 3) << 4;  // staging col {0,16,32,48}

  const _Float16* gK = k16 + plane + (size_t)srr * HD + scc;        // +kp0*HD
  const _Float16* gV = v16t + ((size_t)bh * HD + srr) * SEQ + scc;  // +kp0

  lds_cp32(&Ks[0][srr][scc], gK);  // stage tile 0
  lds_cp32(&Vt[0][srr][scc], gV);

  for (int kt = 0; kt < nkt; ++kt) {
    const int bi = kt & 1;
    // branch-free prefetch of next tile (clamped on the last iteration)
    const int kp1 = ((kt + 1 < nkt) ? (kt + 1) : kt) * 64;
    lds_cp32(&Ks[bi ^ 1][srr][scc], gK + (size_t)kp1 * HD);
    lds_cp32(&Vt[bi ^ 1][srr][scc], gV + kp1);
    ASYNC_WAIT_TILE();
    __syncthreads();

    // ---- S = Q K^T  (16x64 per wave, 2 WMMA per 16x16 tile over hd=64) ---
    float8 s[4];
#pragma unroll
    for (int j = 0; j < 4; ++j) {
      const half16 kb0 = frag_ld(&Ks[bi][j * 16][0], FLD, lane, 0);
      const half16 kb1 = frag_ld(&Ks[bi][j * 16][0], FLD, lane, 32);
      float8 z = {};
      z = WMMA_F32_F16(qf0, kb0, z);
      z = WMMA_F32_F16(qf1, kb1, z);
      s[j] = z;
    }

    // ---- causal mask ------------------------------------------------------
    const int kp0  = kt * 64;
    const int colb = kp0 + (lane & 15);
    const int rb   = qrow0 + ((lane >> 4) << 3);
#pragma unroll
    for (int j = 0; j < 4; ++j)
#pragma unroll
      for (int v = 0; v < 8; ++v)
        if (colb + j * 16 > rb + v) s[j][v] = -1e30f;

    // ---- online softmax (xor masks 1..8 stay inside the 16-lane half) ----
#pragma unroll
    for (int v = 0; v < 8; ++v) {
      float x = fmaxf(fmaxf(s[0][v], s[1][v]), fmaxf(s[2][v], s[3][v]));
      x = fmaxf(x, __shfl_xor(x, 1));
      x = fmaxf(x, __shfl_xor(x, 2));
      x = fmaxf(x, __shfl_xor(x, 4));
      x = fmaxf(x, __shfl_xor(x, 8));
      const float mnew  = fmaxf(mrow[v], x);
      const float alpha = __expf(mrow[v] - mnew);
      mrow[v] = mnew;
      float rs = 0.0f;
#pragma unroll
      for (int j = 0; j < 4; ++j) {
        const float e = __expf(s[j][v] - mnew);
        s[j][v] = e;
        rs += e;
      }
      rs += __shfl_xor(rs, 1);
      rs += __shfl_xor(rs, 2);
      rs += __shfl_xor(rs, 4);
      rs += __shfl_xor(rs, 8);
      lrow[v] = lrow[v] * alpha + rs;
#pragma unroll
      for (int jd = 0; jd < 4; ++jd) o[jd][v] *= alpha;
    }

    // ---- P: C-layout -> A-layout via per-wave LDS (same-wave DS in-order) -
    {
      const int pr = (lane >> 4) << 3;
      const int pc = lane & 15;
#pragma unroll
      for (int j = 0; j < 4; ++j)
#pragma unroll
        for (int v = 0; v < 8; ++v)
          Ps[wave][pr + v][j * 16 + pc] = (_Float16)s[j][v];
    }
    const half16 pf0 = frag_ld(&Ps[wave][0][0], FLD, lane, 0);
    const half16 pf1 = frag_ld(&Ps[wave][0][0], FLD, lane, 32);

    // ---- O += P V ----------------------------------------------------------
#pragma unroll
    for (int jd = 0; jd < 4; ++jd) {
      const half16 vb0 = frag_ld(&Vt[bi][jd * 16][0], FLD, lane, 0);
      const half16 vb1 = frag_ld(&Vt[bi][jd * 16][0], FLD, lane, 32);
      o[jd] = WMMA_F32_F16(pf0, vb0, o[jd]);
      o[jd] = WMMA_F32_F16(pf1, vb1, o[jd]);
    }
    __syncthreads();  // all waves done with buffer bi before it is re-staged
  }

  // ---- normalize, write f16 [B,S,H] for the output projection ------------
#pragma unroll
  for (int v = 0; v < 8; ++v) {
    const float inv = 1.0f / lrow[v];
    const int row = qrow0 + ((lane >> 4) << 3) + v;
#pragma unroll
    for (int jd = 0; jd < 4; ++jd) {
      const int col = h * HD + jd * 16 + (lane & 15);
      attn16[((size_t)b * SEQ + row) * HID + col] = (_Float16)(o[jd][v] * inv);
    }
  }
}

// ---------------------------------------------------------------------------
// Host-side orchestration. Workspace (88 MiB), with aliasing:
//   x16 (8M) | w16[4] (8M) | qf,kf,vf fp32 (48M) | q16,k16,v16 (24M)
//   v16t aliases qf (dead after rope), attn16 aliases kf (dead after rope).
// d_out: y (4.19M f32) ++ k_out (4.19M) ++ v_out (4.19M)
// ---------------------------------------------------------------------------
extern "C" void kernel_launch(void* const* d_in, const int* in_sizes, int n_in,
                              void* d_out, int out_size, void* d_ws,
                              size_t ws_size, hipStream_t stream) {
  const float* x   = (const float*)d_in[0];
  const int*   pos = (const int*)d_in[1];
  const float* Wq  = (const float*)d_in[2];
  const float* Wk  = (const float*)d_in[3];
  const float* Wv  = (const float*)d_in[4];
  const float* Wo  = (const float*)d_in[5];

  float* y     = (float*)d_out;
  float* k_out = y + PLANE;
  float* v_out = y + 2 * PLANE;

  _Float16* x16  = (_Float16*)d_ws;
  _Float16* wq16 = x16 + PLANE;
  _Float16* wk16 = wq16 + (size_t)HID * HID;
  _Float16* wv16 = wk16 + (size_t)HID * HID;
  _Float16* wo16 = wv16 + (size_t)HID * HID;
  float* qf = (float*)(wo16 + (size_t)HID * HID);
  float* kf = qf + PLANE;
  float* vf = kf + PLANE;
  _Float16* q16 = (_Float16*)(vf + PLANE);
  _Float16* k16 = q16 + PLANE;
  _Float16* v16 = k16 + PLANE;
  _Float16* v16t   = (_Float16*)qf;  // alias: qf dead after rope_pack
  _Float16* attn16 = (_Float16*)kf;  // alias: kf dead after rope_pack

  const dim3 blk(256);
  const dim3 ggemm(HID / GBN, MROWS / GBM);  // (8, 32)
  const dim3 gfa(SEQ / 128, BATCH * NH);     // (16, 32)
  const dim3 gtr(SEQ / 64, BATCH * NH);      // (32, 32)

  cvt_f32_f16<<<(int)(PLANE / 1024), blk, 0, stream>>>(x, x16);
  cvt_f32_f16<<<(HID * HID) / 1024, blk, 0, stream>>>(Wq, wq16);
  cvt_f32_f16<<<(HID * HID) / 1024, blk, 0, stream>>>(Wk, wk16);
  cvt_f32_f16<<<(HID * HID) / 1024, blk, 0, stream>>>(Wv, wv16);
  cvt_f32_f16<<<(HID * HID) / 1024, blk, 0, stream>>>(Wo, wo16);

  gemm16<<<ggemm, blk, 0, stream>>>(x16, wq16, qf, MROWS, HID, HID);
  gemm16<<<ggemm, blk, 0, stream>>>(x16, wk16, kf, MROWS, HID, HID);
  gemm16<<<ggemm, blk, 0, stream>>>(x16, wv16, vf, MROWS, HID, HID);

  rope_pack<<<(int)(PLANE / 256), blk, 0, stream>>>(qf, kf, vf, pos, q16, k16,
                                                    v16, k_out, v_out);
  transpose_hd_seq<<<gtr, blk, 0, stream>>>(v16, v16t);
  flash_attn<<<gfa, blk, 0, stream>>>(q16, k16, v16t, attn16);
  gemm16<<<ggemm, blk, 0, stream>>>(attn16, wo16, y, MROWS, HID, HID);
}
